// NeuralAttention_39204461477920
// MI455X (gfx1250) — matrix operand order repaired and checked by reading, not compiled, not run on hardware
//
#include <hip/hip_runtime.h>
#include <hip/hip_bf16.h>

// Sizes from the reference
#define DK   64
#define NB   2
#define NH   8
#define NL   256
#define NF   256          // 4*DK
#define NROWS (NB*NH*NL)  // 4096 "bhl" rows

typedef __attribute__((ext_vector_type(16))) _Float16 v16h;
typedef __attribute__((ext_vector_type(8)))  float    v8f;

// Workspace layout:
//   [0, 64KB)            : W1 packed as WMMA-B register image (f16)
//   [64KB, 64KB+4MB)     : Aq (f32, 4096 x 256)
//   [.., +4MB)           : Ak (f32, 4096 x 256)

// ---------------------------------------------------------------------------
// Kernel 0: pack W1 into the exact per-lane V_WMMA B-operand register image.
// Operand image index (h, kc, nt, lane, e):  element e of lane `lane` holds
//   K = kc*32 + (lane/16)*16 + e,  column f = nt*16 + (lane%16)
// so kernel 1 can fetch a whole v16h with one coalesced 32B load.
// ---------------------------------------------------------------------------
__global__ __launch_bounds__(256)
void na_pack_w1(const float* __restrict__ W1, _Float16* __restrict__ W1h) {
  const int idx  = blockIdx.x * 256 + threadIdx.x;   // 0..2047
  const int lane = idx & 31;
  const int nt   = (idx >> 5) & 15;
  const int kc   = (idx >> 9) & 1;
  const int h    = (idx >> 10) & 1;
  const int col  = nt * 16 + (lane & 15);
  const int kbase = h * 64 + kc * 32 + (lane >> 4) * 16;
  _Float16* dst = W1h + (size_t)idx * 16;
  #pragma unroll
  for (int e = 0; e < 16; ++e)
    dst[e] = (_Float16)W1[(size_t)(kbase + e) * NF + col];
}

// ---------------------------------------------------------------------------
// Kernel 1: projections via WMMA (software-pipelined).
//   wave 0: Aq[16-row tile] = Q_tile @ W1[0:64,:] + b1
//   wave 1: Ak[16-row tile] = K_tile @ W1[64:128,:]
// A is loaded once (4x b128 per K-chunk), B operands stream through a
// depth-2 pipeline so the v_wmma chain never stalls on loadcnt.
// ---------------------------------------------------------------------------
__global__ __launch_bounds__(64)
void na_proj_wmma(const float* __restrict__ Q, const float* __restrict__ K,
                  const _Float16* __restrict__ W1h, const float* __restrict__ b1,
                  float* __restrict__ Aq, float* __restrict__ Ak) {
#if defined(__HIP_DEVICE_COMPILE__)
  const int lane = threadIdx.x & 31;
  const int wave = threadIdx.x >> 5;        // 0 = Q-half, 1 = K-half
  const int tile = blockIdx.x;              // 0..255 -> rows 16*tile..
  const int m    = lane & 15;
  const int hk   = (lane >> 4) & 1;

  const float* X  = wave ? K : Q;
  float*      Out = wave ? Ak : Aq;
  const int rowbase = tile * 16;

  // A operands: contiguous k order, k = 16*hk + e: one 16-float run/lane (matches B pack).
  const float4* arow = (const float4*)(X + (size_t)(rowbase + m) * DK);
  v16h a[2];
  #pragma unroll
  for (int kc = 0; kc < 2; ++kc) {
    const float4 u0 = arow[kc * 8 + hk * 4];
    const float4 u1 = arow[kc * 8 + hk * 4 + 1];
    const float4 u2 = arow[kc * 8 + hk * 4 + 2];
    const float4 u3 = arow[kc * 8 + hk * 4 + 3];
    a[kc][0]  = (_Float16)u0.x; a[kc][1]  = (_Float16)u0.y;
    a[kc][2]  = (_Float16)u0.z; a[kc][3]  = (_Float16)u0.w;
    a[kc][4]  = (_Float16)u1.x; a[kc][5]  = (_Float16)u1.y;
    a[kc][6]  = (_Float16)u1.z; a[kc][7]  = (_Float16)u1.w;
    a[kc][8]  = (_Float16)u2.x; a[kc][9]  = (_Float16)u2.y;
    a[kc][10] = (_Float16)u2.z; a[kc][11] = (_Float16)u2.w;
    a[kc][12] = (_Float16)u3.x; a[kc][13] = (_Float16)u3.y;
    a[kc][14] = (_Float16)u3.z; a[kc][15] = (_Float16)u3.w;
  }

  // Per-wave bases so the unrolled loop becomes immediate-offset addressing.
  const v16h* Bq = (const v16h*)W1h + (size_t)(wave * 2) * 16 * 32 + lane;
  const float bsel = (wave == 0) ? 1.0f : 0.0f;    // branchless bias select
  const float* b1p = b1 + m;
  float* OutBase = Out + (size_t)(rowbase + hk * 8) * NF + m;

  // Depth-2 pipeline: B for nt+1 in flight while WMMAs for nt execute.
  v16h bn0 = Bq[0];          // (kc=0, nt=0)
  v16h bn1 = Bq[16 * 32];    // (kc=1, nt=0)
  #pragma unroll
  for (int nt = 0; nt < 16; ++nt) {
    const v16h c0 = bn0;
    const v16h c1 = bn1;
    if (nt < 15) {
      bn0 = Bq[(size_t)(nt + 1) * 32];
      bn1 = Bq[(size_t)(16 * 32 + (nt + 1) * 32)];
    }
    v8f acc = {};
    acc = __builtin_amdgcn_wmma_f32_16x16x32_f16(
        false, a[0], false, c0, (short)0, acc, false, false);
    acc = __builtin_amdgcn_wmma_f32_16x16x32_f16(
        false, a[1], false, c1, (short)0, acc, false, false);

    const float bv = b1p[nt * 16] * bsel;
    #pragma unroll
    for (int r = 0; r < 8; ++r)
      OutBase[r * NF + nt * 16] = acc[r] + bv;
  }
#endif
}

// ---------------------------------------------------------------------------
// Kernel 2: pairwise relu-dot.  One 256-thread block per (bh, i-tile, j-tile).
// LDS staging uses CDNA5 GLOBAL_LOAD_ASYNC_TO_LDS_B128 (ASYNCcnt-tracked,
// bypasses VGPRs), then s_wait_asynccnt 0 + barrier before compute.
// ---------------------------------------------------------------------------
#define ROWP (NF + 4)   // padded LDS row stride (floats)

__global__ __launch_bounds__(256)
void na_pair_relu_dot(const float* __restrict__ Aq, const float* __restrict__ Ak,
                      const float* __restrict__ W2, const float* __restrict__ b2,
                      float* __restrict__ out) {
#if defined(__HIP_DEVICE_COMPILE__)
  __shared__ float aq_s[16 * ROWP];
  __shared__ float ak_s[16 * ROWP];
  __shared__ float w2_s[NF];

  const int tid = threadIdx.x;      // 0..255
  const int bh  = blockIdx.z;
  const int it  = blockIdx.y;
  const int jt  = blockIdx.x;

  // Async DMA fill: thread t stages 16 floats of each tile (4x b128 each).
  const int lrow = tid >> 4;
  const int lcol = (tid & 15) * 16;
  {
    const unsigned long long aq_g =
        (unsigned long long)(uintptr_t)(Aq + ((size_t)(bh * NL + it * 16 + lrow)) * NF + lcol);
    const unsigned long long ak_g =
        (unsigned long long)(uintptr_t)(Ak + ((size_t)(bh * NL + jt * 16 + lrow)) * NF + lcol);
    const unsigned aq_l = (unsigned)(uintptr_t)&aq_s[lrow * ROWP + lcol];
    const unsigned ak_l = (unsigned)(uintptr_t)&ak_s[lrow * ROWP + lcol];
    #pragma unroll
    for (int q = 0; q < 4; ++q) {
      asm volatile("global_load_async_to_lds_b128 %0, %1, off"
                   :: "v"(aq_l + 16u * q), "v"(aq_g + 16u * q) : "memory");
      asm volatile("global_load_async_to_lds_b128 %0, %1, off"
                   :: "v"(ak_l + 16u * q), "v"(ak_g + 16u * q) : "memory");
    }
    if (tid < NF / 4) {
      const unsigned long long w2_g =
          (unsigned long long)(uintptr_t)(W2 + tid * 4);
      const unsigned w2_l = (unsigned)(uintptr_t)&w2_s[tid * 4];
      asm volatile("global_load_async_to_lds_b128 %0, %1, off"
                   :: "v"(w2_l), "v"(w2_g) : "memory");
    }
  }
  asm volatile("s_wait_asynccnt 0x0" ::: "memory");
  __syncthreads();

  const int i = tid >> 4;
  const int j = tid & 15;
  const float* ar = &aq_s[i * ROWP];
  const float* kr = &ak_s[j * ROWP];

  float s = 0.0f;
  #pragma unroll 4
  for (int f = 0; f < NF; f += 4) {
    const float4 av = *(const float4*)&ar[f];
    const float4 kv = *(const float4*)&kr[f];
    const float4 wv = *(const float4*)&w2_s[f];
    s = fmaf(fmaxf(av.x + kv.x, 0.0f), wv.x, s);
    s = fmaf(fmaxf(av.y + kv.y, 0.0f), wv.y, s);
    s = fmaf(fmaxf(av.z + kv.z, 0.0f), wv.z, s);
    s = fmaf(fmaxf(av.w + kv.w, 0.0f), wv.w, s);
  }

  out[((size_t)(bh * NL + it * 16 + i)) * NL + jt * 16 + j] = s + b2[0];
#endif
}

// ---------------------------------------------------------------------------
extern "C" void kernel_launch(void* const* d_in, const int* in_sizes, int n_in,
                              void* d_out, int out_size, void* d_ws, size_t ws_size,
                              hipStream_t stream) {
  (void)in_sizes; (void)n_in; (void)out_size; (void)ws_size;
  const float* Q  = (const float*)d_in[0];
  const float* K  = (const float*)d_in[1];
  const float* W1 = (const float*)d_in[2];
  const float* b1 = (const float*)d_in[3];
  const float* W2 = (const float*)d_in[4];
  const float* b2 = (const float*)d_in[5];
  float* out = (float*)d_out;

  _Float16* W1h = (_Float16*)d_ws;                          // 64 KB
  float* Aq = (float*)((char*)d_ws + 64 * 1024);            // 4 MB
  float* Ak = Aq + (size_t)NROWS * NF;                      // 4 MB

  // Phase 0: pack W1 into WMMA-B register image (one-time reshape, reused by
  // all 512 projection waves).
  na_pack_w1<<<dim3(8), dim3(256), 0, stream>>>(W1, W1h);

  // Phase 1: WMMA projections (256 blocks x 2 waves).
  na_proj_wmma<<<dim3(NROWS / 16), dim3(64), 0, stream>>>(Q, K, W1h, b1, Aq, Ak);

  // Phase 2: pairwise relu-dot (16 x 16 x 16 tiles, 256 threads each).
  na_pair_relu_dot<<<dim3(16, 16, 16), dim3(256), 0, stream>>>(Aq, Ak, W2, b2, out);
}
